// CompressionGainAnalyzer_59614146069049
// MI455X (gfx1250) — matrix operand order repaired and checked by reading, not compiled, not run on hardware
//
#include <hip/hip_runtime.h>
#include <math.h>

// ---------------- problem constants (match reference) ----------------
#define BN   131072
#define DD   512
#define HH   256
#define ZZ   128
#define KK   256
// main-kernel tiling
#define ROWS_PER_WG 64          // 4 waves * 16 rows
#define WG_THREADS  128
#define N_WG        (BN / ROWS_PER_WG)   // 2048

typedef __attribute__((ext_vector_type(16))) __bf16 bf16x16;
typedef __attribute__((ext_vector_type(8)))  float  f32x8;

union Frag { int4 q[2]; bf16x16 v; };

static __device__ __forceinline__ unsigned short f2bf(float f) {
    unsigned int u = __float_as_uint(f);
    unsigned int r = (u + 0x7fffu + ((u >> 16) & 1u)) >> 16;   // RNE
    return (unsigned short)r;
}
static __device__ __forceinline__ float bf2f(unsigned int h) {
    return __uint_as_float((h & 0xffffu) << 16);
}
static __device__ __forceinline__ float gelu_exact(float x) {
    return 0.5f * x * (1.0f + erff(x * 0.70710678118654752f));
}

// ---------------------------------------------------------------------
// Prologue 1: repack a [Kd x Nd] f32 weight into WMMA-B fragment order
// (bf16). Per 32x16 tile: lane = ((kk>>3)&1)*16 + nn holds 16 bf16
// (slot = (kk&7) | (kk>>4)<<3), so the hot loop loads 2 x b128 per lane.
// transpose=1 reads W as [Nd x Kd] (for codebook^T).
// ---------------------------------------------------------------------
__global__ void k_prep_bfrag(const float* __restrict__ W,
                             unsigned short* __restrict__ out,
                             int Kd, int Nd, int transpose) {
    int idx = blockIdx.x * blockDim.x + threadIdx.x;
    if (idx >= Kd * Nd) return;
    int k = idx / Nd, n = idx % Nd;
    int kt = k >> 5, kk = k & 31, nt = n >> 4, nn = n & 15;
    int lane = (((kk >> 3) & 1) << 4) | nn;
    int slot = (kk & 7) | ((kk >> 4) << 3);
    int NT = Nd >> 4;
    size_t o = ((size_t)(kt * NT + nt) * 32 + lane) * 16 + slot;
    float v = transpose ? W[(size_t)n * Kd + k] : W[(size_t)k * Nd + n];
    out[o] = f2bf(v);
}

// Prologue 2: half squared norms of codebook rows
__global__ void k_cnorm(const float* __restrict__ cb, float* __restrict__ cn) {
    int n = threadIdx.x;
    float s = 0.f;
    for (int z = 0; z < ZZ; ++z) { float v = cb[n * ZZ + z]; s += v * v; }
    cn[n] = 0.5f * s;
}

// ---------------------------------------------------------------------
// Prologue 3: decoded_codebook[n] = decoder(codebook[n])  (256 x 512 f32)
// one block per codebook entry
// ---------------------------------------------------------------------
__global__ __launch_bounds__(256) void k_decode(
    const float* __restrict__ cb,
    const float* __restrict__ w1, const float* __restrict__ b1,
    const float* __restrict__ g,  const float* __restrict__ be,
    const float* __restrict__ w2, const float* __restrict__ b2,
    float* __restrict__ dc) {
    __shared__ float q[ZZ];
    __shared__ float h[HH];
    __shared__ float red[HH];
    int t = threadIdx.x, n = blockIdx.x;
    if (t < ZZ) q[t] = cb[n * ZZ + t];
    __syncthreads();
    float s = b1[t];
    for (int z = 0; z < ZZ; ++z) s += q[z] * w1[z * HH + t];
    red[t] = s; __syncthreads();
    for (int off = 128; off > 0; off >>= 1) { if (t < off) red[t] += red[t + off]; __syncthreads(); }
    float mu = red[0] * (1.0f / HH); __syncthreads();
    float d = s - mu;
    red[t] = d * d; __syncthreads();
    for (int off = 128; off > 0; off >>= 1) { if (t < off) red[t] += red[t + off]; __syncthreads(); }
    float var = red[0] * (1.0f / HH); __syncthreads();
    float hv = d * rsqrtf(var + 1e-5f) * g[t] + be[t];
    h[t] = gelu_exact(hv); __syncthreads();
    for (int c = t; c < DD; c += 256) {
        float sv = b2[c];
        for (int j = 0; j < HH; ++j) sv += h[j] * w2[j * DD + c];
        dc[(size_t)n * DD + c] = sv;
    }
}

// ---------------------------------------------------------------------
// Main fused kernel: encoder (WMMA bf16) -> argmin over codebook (WMMA)
// -> reconstruction error via decoded-codebook lookup.
// 4 waves x 16 rows, wave-local after bias staging up to the per-block error sum.
// ---------------------------------------------------------------------
__global__ __launch_bounds__(WG_THREADS) void k_main(
    const float* __restrict__ feat,
    const float* __restrict__ enc_b1, const float* __restrict__ enc_g,
    const float* __restrict__ enc_be, const float* __restrict__ enc_b2,
    const unsigned short* __restrict__ w1f,
    const unsigned short* __restrict__ w2f,
    const unsigned short* __restrict__ cbtf,
    const float* __restrict__ cnorm,
    const float* __restrict__ dc,
    float* __restrict__ out_err, float* __restrict__ out_idx,
    float* __restrict__ partials) {
    extern __shared__ __align__(16) char smem[];
    unsigned short* Xbf = (unsigned short*)smem;                 // 64*512
    unsigned short* act = Xbf + ROWS_PER_WG * DD;                // 64*256
    unsigned short* Ebf = act + ROWS_PER_WG * HH;                // 64*128
    float* sb1 = (float*)(Ebf + ROWS_PER_WG * ZZ);               // 256
    float* sg  = sb1 + HH;                                       // 256
    float* sbe = sg + HH;                                        // 256
    float* sb2 = sbe + HH;                                       // 128
    int*   sidx = (int*)(sb2 + ZZ);                              // 64
    float* swsum = (float*)(sidx + ROWS_PER_WG);                 // 4

    const int tid  = threadIdx.x;
    const int lane = tid & 31;
    const int w    = tid >> 5;        // wave id 0..3
    const int nn   = lane & 15;
    const int half = lane >> 4;
    const int row0 = blockIdx.x * ROWS_PER_WG;   // global base row
    const int wr   = w * 16;                     // wave local base row

    // stage small bias vectors once (one of two block-wide syncs in the kernel body)
    for (int i = tid; i < HH; i += WG_THREADS) { sb1[i] = enc_b1[i]; sg[i] = enc_g[i]; sbe[i] = enc_be[i]; }
    if (tid < ZZ) sb2[tid] = enc_b2[tid];
    __syncthreads();

    // ---- load this wave's 16x512 feature rows, convert f32 -> bf16 ----
    {
        const float4* src = (const float4*)(feat + (size_t)(row0 + wr) * DD);
        uint2* dst = (uint2*)(Xbf + wr * DD);
        for (int e = lane; e < 16 * DD / 4; e += 32) {
            float4 v = src[e];
            uint2 p;
            p.x = (unsigned)f2bf(v.x) | ((unsigned)f2bf(v.y) << 16);
            p.y = (unsigned)f2bf(v.z) | ((unsigned)f2bf(v.w) << 16);
            dst[e] = p;
        }
    }

    const int arow = wr + nn;   // A-fragment row owned by this lane

    // ---- GEMM1: H1 = X @ enc_w1   (16x512 @ 512x256) ----
    for (int nt = 0; nt < 16; ++nt) {
        f32x8 acc = {};
        for (int kt = 0; kt < 16; ++kt) {
            Frag a, b;
            a.q[0] = *(const int4*)(Xbf + arow * DD + kt * 32 + half * 8);
            a.q[1] = *(const int4*)(Xbf + arow * DD + kt * 32 + 16 + half * 8);
            const int4* bp = (const int4*)(w1f + ((size_t)(kt * 16 + nt) * 32 + lane) * 16);
            b.q[0] = bp[0]; b.q[1] = bp[1];
            acc = __builtin_amdgcn_wmma_f32_16x16x32_bf16(false, a.v, false, b.v,
                                                          (short)0, acc, false, false);
        }
        int col = nt * 16 + nn;
#pragma unroll
        for (int i = 0; i < 8; ++i)
            act[(wr + half * 8 + i) * HH + col] = f2bf(acc[i]);
    }

    // ---- LayerNorm(+bias) + exact GELU, in place (2 lanes per row) ----
    {
        int row = wr + nn;
        int c0 = half * 128;
        float s = 0.f, s2 = 0.f;
        for (int c = 0; c < 128; ++c) {
            float v = bf2f(act[row * HH + c0 + c]) + sb1[c0 + c];
            s += v; s2 += v * v;
        }
        s  += __shfl_xor(s, 16, 32);
        s2 += __shfl_xor(s2, 16, 32);
        float mu  = s * (1.0f / HH);
        float var = s2 * (1.0f / HH) - mu * mu;
        float inv = rsqrtf(var + 1e-5f);
        for (int c = 0; c < 128; ++c) {
            int cc = c0 + c;
            float v = bf2f(act[row * HH + cc]) + sb1[cc];
            float hn = (v - mu) * inv * sg[cc] + sbe[cc];
            act[row * HH + cc] = f2bf(gelu_exact(hn));
        }
    }

    // ---- GEMM2: E = H1g @ enc_w2 + b2   (16x256 @ 256x128) ----
    for (int nt = 0; nt < 8; ++nt) {
        f32x8 acc = {};
        for (int kt = 0; kt < 8; ++kt) {
            Frag a, b;
            a.q[0] = *(const int4*)(act + arow * HH + kt * 32 + half * 8);
            a.q[1] = *(const int4*)(act + arow * HH + kt * 32 + 16 + half * 8);
            const int4* bp = (const int4*)(w2f + ((size_t)(kt * 8 + nt) * 32 + lane) * 16);
            b.q[0] = bp[0]; b.q[1] = bp[1];
            acc = __builtin_amdgcn_wmma_f32_16x16x32_bf16(false, a.v, false, b.v,
                                                          (short)0, acc, false, false);
        }
        int col = nt * 16 + nn;
        float bias = sb2[col];
#pragma unroll
        for (int i = 0; i < 8; ++i)
            Ebf[(wr + half * 8 + i) * ZZ + col] = f2bf(acc[i] + bias);
    }

    // ---- GEMM3: S = E @ codebook^T; argmin of 0.5||c||^2 - S ----
    float bv[8]; int bi[8];
#pragma unroll
    for (int i = 0; i < 8; ++i) { bv[i] = 3.4e38f; bi[i] = 0; }
    for (int nt = 0; nt < 16; ++nt) {
        f32x8 acc = {};
        for (int kt = 0; kt < 4; ++kt) {
            Frag a, b;
            a.q[0] = *(const int4*)(Ebf + arow * ZZ + kt * 32 + half * 8);
            a.q[1] = *(const int4*)(Ebf + arow * ZZ + kt * 32 + 16 + half * 8);
            const int4* bp = (const int4*)(cbtf + ((size_t)(kt * 16 + nt) * 32 + lane) * 16);
            b.q[0] = bp[0]; b.q[1] = bp[1];
            acc = __builtin_amdgcn_wmma_f32_16x16x32_bf16(false, a.v, false, b.v,
                                                          (short)0, acc, false, false);
        }
        int n = nt * 16 + nn;
        float cn = cnorm[n];
#pragma unroll
        for (int i = 0; i < 8; ++i) {
            float v = cn - acc[i];
            if (v < bv[i]) { bv[i] = v; bi[i] = n; }   // ascending n => first-min kept
        }
    }
    // butterfly min across the 16 lanes that share each row
#pragma unroll
    for (int i = 0; i < 8; ++i) {
        float v = bv[i]; int ix = bi[i];
        for (int m = 1; m < 16; m <<= 1) {
            float ov = __shfl_xor(v, m, 32);
            int   oi = __shfl_xor(ix, m, 32);
            if (ov < v || (ov == v && oi < ix)) { v = ov; ix = oi; }
        }
        if (nn == i) {                         // one writer per half
            int lrow = wr + half * 8 + i;
            sidx[lrow] = ix;
            out_idx[row0 + lrow] = (float)ix;
        }
    }

    // ---- reconstruction error via decoded-codebook lookup ----
    {
        int lrow = wr + nn;
        int ridx = sidx[lrow];
        const float4* dcr = (const float4*)(dc + (size_t)ridx * DD + half * 256);
        const uint2*  xr  = (const uint2*)(Xbf + lrow * DD + half * 256);
        float s = 0.f;
        for (int e = 0; e < 64; ++e) {
            float4 dv = dcr[e];
            uint2  xv = xr[e];
            float d0 = dv.x - bf2f(xv.x);
            float d1 = dv.y - bf2f(xv.x >> 16);
            float d2 = dv.z - bf2f(xv.y);
            float d3 = dv.w - bf2f(xv.y >> 16);
            s += d0 * d0 + d1 * d1 + d2 * d2 + d3 * d3;
        }
        s += __shfl_xor(s, 16, 32);
        float err = s * (1.0f / DD);
        if (half == 0) out_err[row0 + lrow] = err;
        float ps = (half == 0) ? err : 0.0f;
        for (int m = 1; m < 32; m <<= 1) ps += __shfl_xor(ps, m, 32);
        if (lane == 0) swsum[w] = ps;
    }
    __syncthreads();
    if (tid == 0)
        partials[blockIdx.x] = swsum[0] + swsum[1] + swsum[2] + swsum[3];
}

// fixed-order deterministic reduction of per-block error sums
__global__ __launch_bounds__(256) void k_sum(const float* __restrict__ p,
                                             float* __restrict__ total, int n) {
    __shared__ float red[256];
    int t = threadIdx.x; float s = 0.f;
    for (int i = t; i < n; i += 256) s += p[i];
    red[t] = s; __syncthreads();
    for (int off = 128; off > 0; off >>= 1) { if (t < off) red[t] += red[t + off]; __syncthreads(); }
    if (t == 0) *total = red[0];
}

// per-row bit estimation & compression ratio
__global__ void k_final(float* __restrict__ out, const float* __restrict__ total) {
    int i = blockIdx.x * blockDim.x + threadIdx.x;
    if (i >= BN) return;
    float scale = (*total) * (1.0f / BN) + 1e-8f;
    float err = out[i];
    float lp = -fabsf(err) / scale - logf(2.0f * scale);
    float eb = -lp * 1.4426950408889634f;   // / ln2
    float tb = 8.0f + eb;                   // log2(256) = 8
    out[(size_t)BN + i]     = (DD * 32.0f) / tb;  // compression_ratio
    out[2 * (size_t)BN + i] = 0.0f;               // compression_gain
    out[3 * (size_t)BN + i] = tb;                 // total_bits
}

extern "C" void kernel_launch(void* const* d_in, const int* in_sizes, int n_in,
                              void* d_out, int out_size, void* d_ws, size_t ws_size,
                              hipStream_t stream) {
    const float* feat   = (const float*)d_in[0];
    const float* enc_w1 = (const float*)d_in[1];
    const float* enc_b1 = (const float*)d_in[2];
    const float* enc_g  = (const float*)d_in[3];
    const float* enc_be = (const float*)d_in[4];
    const float* enc_w2 = (const float*)d_in[5];
    const float* enc_b2 = (const float*)d_in[6];
    const float* cb     = (const float*)d_in[7];
    const float* dec_w1 = (const float*)d_in[8];
    const float* dec_b1 = (const float*)d_in[9];
    const float* dec_g  = (const float*)d_in[10];
    const float* dec_be = (const float*)d_in[11];
    const float* dec_w2 = (const float*)d_in[12];
    const float* dec_b2 = (const float*)d_in[13];
    (void)in_sizes; (void)n_in; (void)out_size; (void)ws_size;

    // workspace layout (bytes)
    char* p = (char*)d_ws;
    unsigned short* w1f   = (unsigned short*)p;              p += (size_t)DD * HH * 2;   // 256KB
    unsigned short* w2f   = (unsigned short*)p;              p += (size_t)HH * ZZ * 2;   // 64KB
    unsigned short* cbtf  = (unsigned short*)p;              p += (size_t)ZZ * KK * 2;   // 64KB
    float*          cnorm = (float*)p;                       p += (size_t)KK * 4;
    float*          dc    = (float*)p;                       p += (size_t)KK * DD * 4;   // 512KB
    float*          parts = (float*)p;                       p += (size_t)N_WG * 4;
    float*          total = (float*)p;

    float* out      = (float*)d_out;
    float* out_err  = out;                 // [B]
    float* out_idx  = out + 4 * (size_t)BN; // quantized_indices as float

    // prologue: weight repack + codebook norms + decoded codebook
    k_prep_bfrag<<<(DD * HH + 255) / 256, 256, 0, stream>>>(enc_w1, w1f, DD, HH, 0);
    k_prep_bfrag<<<(HH * ZZ + 255) / 256, 256, 0, stream>>>(enc_w2, w2f, HH, ZZ, 0);
    k_prep_bfrag<<<(ZZ * KK + 255) / 256, 256, 0, stream>>>(cb, cbtf, ZZ, KK, 1);
    k_cnorm<<<1, KK, 0, stream>>>(cb, cnorm);
    k_decode<<<KK, 256, 0, stream>>>(cb, dec_w1, dec_b1, dec_g, dec_be, dec_w2, dec_b2, dc);

    // main fused encoder + VQ + error kernel
    size_t lds = (size_t)ROWS_PER_WG * DD * 2     // Xbf
               + (size_t)ROWS_PER_WG * HH * 2     // act
               + (size_t)ROWS_PER_WG * ZZ * 2     // Ebf
               + (3 * HH + ZZ) * 4                // biases/gains
               + ROWS_PER_WG * 4                  // idx
               + 16;                              // wave sums
    k_main<<<N_WG, WG_THREADS, lds, stream>>>(feat, enc_b1, enc_g, enc_be, enc_b2,
                                              w1f, w2f, cbtf, cnorm, dc,
                                              out_err, out_idx, parts);

    // deterministic scale + per-row bits
    k_sum<<<1, 256, 0, stream>>>(parts, total, N_WG);
    k_final<<<(BN + 255) / 256, 256, 0, stream>>>(out, total);
}
